// MultiHeadSelfAttention_32667521254206
// MI455X (gfx1250) — matrix operand order repaired and checked by reading, not compiled, not run on hardware
//
#include <hip/hip_runtime.h>

typedef __attribute__((ext_vector_type(16))) __bf16 v16bf;
typedef __attribute__((ext_vector_type(8)))  float  v8f;
typedef unsigned short u16;
typedef unsigned int   u32;

// ---------- helpers ----------
__device__ inline u16 f2bf(float f) {
  union { float f; u32 u; } v; v.f = f;
  u32 r = v.u + 0x7fffu + ((v.u >> 16) & 1u);
  return (u16)(r >> 16);
}
__device__ inline float bf2f(u16 h) {
  union { u32 u; float f; } v; v.u = ((u32)h) << 16; return v.f;
}

union FragA { v16bf m; u32 w[8]; };
union FragC { v8f  m; float f[8]; };

__device__ inline v8f wmma_bf16(v16bf a, v16bf b, v8f c) {
  // (neg_a, A, neg_b, B, c_mod, C, reuse_a, reuse_b)
  return __builtin_amdgcn_wmma_f32_16x16x32_bf16(false, a, false, b, (short)0, c, false, false);
}

// gfx1250 async global->LDS copy (16 bytes per active lane), ASYNCcnt-tracked
__device__ inline void async_ld_b128(void* lds, const void* g) {
  u32 l = (u32)(uintptr_t)lds;  // low 32 bits of generic shared address = LDS offset
  asm volatile("global_load_async_to_lds_b128 %0, %1, off" :: "v"(l), "v"(g) : "memory");
}
__device__ inline void wait_async() {
  asm volatile("s_wait_asynccnt 0" ::: "memory");
}

// ---- 16-lane reductions via DPP16 (pure VALU, halves of wave32 never mix) ----
// quad_perm xor1 = 0xB1, quad_perm xor2 = 0x4E, ROW_HALF_MIRROR = 0x141, ROW_MIRROR = 0x140
template <int CTRL>
__device__ inline float dpp_bcast(float x) {
  int xi = __builtin_bit_cast(int, x);
  int yi = __builtin_amdgcn_update_dpp(0, xi, CTRL, 0xf, 0xf, true);
  return __builtin_bit_cast(float, yi);
}
__device__ inline float halfmax16(float x) {
  x = fmaxf(x, dpp_bcast<0xB1>(x));
  x = fmaxf(x, dpp_bcast<0x4E>(x));
  x = fmaxf(x, dpp_bcast<0x141>(x));
  x = fmaxf(x, dpp_bcast<0x140>(x));
  return x;
}
__device__ inline float halfsum16(float x) {
  x += dpp_bcast<0xB1>(x);
  x += dpp_bcast<0x4E>(x);
  x += dpp_bcast<0x141>(x);
  x += dpp_bcast<0x140>(x);
  return x;
}

// ---------- fp32 -> bf16 conversion (vectorized x4) ----------
__global__ __launch_bounds__(256) void cvt_f32_bf16(const float4* __restrict__ in,
                                                    uint2* __restrict__ out, int n4) {
  int i = blockIdx.x * blockDim.x + threadIdx.x;
  if (i >= n4) return;
  float4 v = in[i];
  uint2 o;
  o.x = (u32)f2bf(v.x) | ((u32)f2bf(v.y) << 16);
  o.y = (u32)f2bf(v.z) | ((u32)f2bf(v.w) << 16);
  out[i] = o;
}

// ---------- RoPE (in-place on bf16 Q and K, layout (B*S, D), head dim dk) ----------
__global__ __launch_bounds__(256) void rope_kernel(u16* __restrict__ Q, u16* __restrict__ K,
                                                   int rows, int S, int D, int dk) {
  int idx = blockIdx.x * blockDim.x + threadIdx.x;
  int halfD = D >> 1;
  if (idx >= rows * halfD) return;
  int row = idx / halfD;
  int pj  = idx - row * halfD;       // pair index within row; col = 2*pj
  int i   = pj & (dk / 2 - 1);       // pair index within head
  int s   = row & (S - 1);
  float ang = (float)s * __powf(10000.0f, -2.0f * (float)i / (float)dk);
  float sn, cs;
  __sincosf(ang, &sn, &cs);
  size_t base = (size_t)row * D + (size_t)(pj << 1);
  {
    float a = bf2f(Q[base]), b = bf2f(Q[base + 1]);
    Q[base]     = f2bf(a * cs - b * sn);
    Q[base + 1] = f2bf(b * cs + a * sn);
  }
  {
    float a = bf2f(K[base]), b = bf2f(K[base + 1]);
    K[base]     = f2bf(a * cs - b * sn);
    K[base + 1] = f2bf(b * cs + a * sn);
  }
}

// ---------- GEMM: C[M,N] = A[M,Kd](bf16) * W[Kd,N](bf16), fp32 accum ----------
// block tile 128x128, BK=64, 8 waves (4x2), wave tile 32x64 (2x4 wmma tiles)
// double-buffered LDS; A staged via async global->LDS, B via regs (transposed)
constexpr int BM = 128, BN = 128, BK = 64;

template <bool OUT_BF16>
__global__ __launch_bounds__(256) void gemm_bf16_kernel(const u16* __restrict__ A,
                                                        const u16* __restrict__ W,
                                                        void* __restrict__ Cout,
                                                        int N, int Kd) {
  __shared__ __align__(16) u16 As[2][BM][BK + 8];   // [buf][row][k]   (stride 144B)
  __shared__ __align__(16) u16 Bs[2][BN][BK + 8];   // [buf][col][k]   (transposed)

  const int tid = threadIdx.x;
  const int wid = tid >> 5;
  const int lane = tid & 31;
  const int hf = lane >> 4;
  const int l16 = lane & 15;
  const int wm = wid >> 1;           // 0..3 -> 32-row band
  const int wn = wid & 1;            // 0..1 -> 64-col band
  const int m0 = blockIdx.y * BM;
  const int n0 = blockIdx.x * BN;

  FragC acc[2][4];
#pragma unroll
  for (int a = 0; a < 2; a++)
#pragma unroll
    for (int b = 0; b < 4; b++)
#pragma unroll
      for (int j = 0; j < 8; j++) acc[a][b].f[j] = 0.0f;

  uint2 breg[2][4];

  // ---- prologue: stage tile 0 ----
#pragma unroll
  for (int j = 0; j < 4; j++) {              // A: 1024 16B chunks
    int ci = tid + j * 256;
    int r = ci >> 3, c = (ci & 7) * 8;
    async_ld_b128(&As[0][r][c], A + (size_t)(m0 + r) * Kd + c);
  }
#pragma unroll
  for (int j = 0; j < 2; j++) {              // B: 512 chunks of 4 cols x 4 k
    int ci = tid + j * 256;
    int kq = ci >> 5, nc = (ci & 31) * 4;
#pragma unroll
    for (int r = 0; r < 4; r++)
      breg[j][r] = *(const uint2*)(W + (size_t)(kq * 4 + r) * N + n0 + nc);
  }
#pragma unroll
  for (int j = 0; j < 2; j++) {
    int ci = tid + j * 256;
    int kq = ci >> 5, nc = (ci & 31) * 4;
#pragma unroll
    for (int c2 = 0; c2 < 4; c2++) {
      uint2 st;
      st.x = (u32)((const u16*)&breg[j][0])[c2] | ((u32)((const u16*)&breg[j][1])[c2] << 16);
      st.y = (u32)((const u16*)&breg[j][2])[c2] | ((u32)((const u16*)&breg[j][3])[c2] << 16);
      *(uint2*)(&Bs[0][nc + c2][kq * 4]) = st;
    }
  }
  wait_async();
  __syncthreads();

  const int nsteps = Kd / BK;
  for (int s = 0; s < nsteps; s++) {
    const int buf = s & 1;
    const int k0n = (s + 1) * BK;
    const bool more = (s + 1) < nsteps;

    // 1) issue async A loads for next tile (no wait)
    if (more) {
#pragma unroll
      for (int j = 0; j < 4; j++) {
        int ci = tid + j * 256;
        int r = ci >> 3, c = (ci & 7) * 8;
        async_ld_b128(&As[buf ^ 1][r][c], A + (size_t)(m0 + r) * Kd + k0n + c);
      }
      // 2) issue global loads of next B tile into registers
#pragma unroll
      for (int j = 0; j < 2; j++) {
        int ci = tid + j * 256;
        int kq = ci >> 5, nc = (ci & 31) * 4;
#pragma unroll
        for (int r = 0; r < 4; r++)
          breg[j][r] = *(const uint2*)(W + (size_t)(k0n + kq * 4 + r) * N + n0 + nc);
      }
      if (s + 2 < nsteps)    // prefetch W tile for step s+2 (stays hot in WGP$/L2)
        __builtin_prefetch(W + (size_t)((s + 2) * BK + (tid & 63)) * N + n0 + (tid >> 6) * 32, 0, 0);
    }

    // 3) compute current tile: preload all fragments, then batch 16 WMMAs
    FragA af[2][2], bfm[2][4];
#pragma unroll
    for (int kc = 0; kc < 2; kc++) {
#pragma unroll
      for (int t = 0; t < 2; t++) {
        int row = wm * 32 + t * 16 + l16;
#pragma unroll
        for (int p = 0; p < 8; p++) {
          int kb = kc * 32 + (p < 4 ? 2 * p : 8 + 2 * p) + hf * 8;  // A frag layout
          af[kc][t].w[p] = *(const u32*)(&As[buf][row][kb]);
        }
      }
#pragma unroll
      for (int t = 0; t < 4; t++) {
        int col = wn * 64 + t * 16 + l16;
#pragma unroll
        for (int p = 0; p < 8; p++) {
          int kb = kc * 32 + (p < 4 ? 2 * p : 8 + 2 * p) + hf * 8;  // B frag layout
          bfm[kc][t].w[p] = *(const u32*)(&Bs[buf][col][kb]);
        }
      }
    }
#pragma unroll
    for (int tm = 0; tm < 2; tm++)
#pragma unroll
      for (int tn = 0; tn < 4; tn++)
#pragma unroll
        for (int kc = 0; kc < 2; kc++)
          acc[tm][tn].m = wmma_bf16(af[kc][tm].m, bfm[kc][tn].m, acc[tm][tn].m);

    // 4) drain staged B into next buffer, 5) wait async + barrier
    if (more) {
#pragma unroll
      for (int j = 0; j < 2; j++) {
        int ci = tid + j * 256;
        int kq = ci >> 5, nc = (ci & 31) * 4;
#pragma unroll
        for (int c2 = 0; c2 < 4; c2++) {
          uint2 st;
          st.x = (u32)((const u16*)&breg[j][0])[c2] | ((u32)((const u16*)&breg[j][1])[c2] << 16);
          st.y = (u32)((const u16*)&breg[j][2])[c2] | ((u32)((const u16*)&breg[j][3])[c2] << 16);
          *(uint2*)(&Bs[buf ^ 1][nc + c2][kq * 4]) = st;
        }
      }
      wait_async();
    }
    __syncthreads();
  }

  // store (C layout: VGPR r -> row r + half*8, col = lane%16)
#pragma unroll
  for (int tm = 0; tm < 2; tm++) {
#pragma unroll
    for (int tn = 0; tn < 4; tn++) {
      int rb = m0 + wm * 32 + tm * 16 + hf * 8;
      int cb = n0 + wn * 64 + tn * 16 + l16;
#pragma unroll
      for (int r = 0; r < 8; r++) {
        float v = acc[tm][tn].f[r];
        size_t off = (size_t)(rb + r) * N + cb;
        if (OUT_BF16) ((u16*)Cout)[off] = f2bf(v);
        else          ((float*)Cout)[off] = v;
      }
    }
  }
}

// ---------- Flash attention: per block = (b, h, 128-row q tile); 8 waves x 16 rows ----------
constexpr int QT = 128;

__global__ __launch_bounds__(256) void attn_kernel(const u16* __restrict__ Q,
                                                   const u16* __restrict__ K,
                                                   const u16* __restrict__ V,
                                                   u16* __restrict__ O,
                                                   int S, int H, int dk) {
  const int D = H * dk;                     // 2048
  const int numQT = S / QT;                 // 16
  const int bh = blockIdx.x / numQT;
  const int qt = blockIdx.x - bh * numQT;
  const int b = bh / H, h = bh - b * H;

  const size_t headOff = ((size_t)b * S) * D + (size_t)h * dk;
  const u16* Qh = Q + headOff;
  const u16* Kh = K + headOff;
  const u16* Vh = V + headOff;
  u16* Oh = O + headOff;

  const int tid = threadIdx.x;
  const int wid = tid >> 5;                 // 0..7
  const int lane = tid & 31;
  const int hf = lane >> 4;
  const int l16 = lane & 15;
  const int q0 = qt * QT;
  const int qw = q0 + wid * 16;

  __shared__ __align__(16) u16 Ks[32][136];     // [key][d]   (stride 272B)
  __shared__ __align__(16) u16 Vs[128][40];     // [d][key]   (transposed, stride 80B)
  __shared__ __align__(16) u16 Ps[8][16][40];   // per-wave P scratch (stride 80B)

  // Q fragments: 4 K-chunks of 32 over dk=128, held for the whole loop
  FragA qf[4];
  {
    const u16* qrow = Qh + (size_t)(qw + l16) * D;
#pragma unroll
    for (int kc = 0; kc < 4; kc++)
#pragma unroll
      for (int p = 0; p < 8; p++) {
        int kb = kc * 32 + (p < 4 ? 2 * p : 8 + 2 * p) + hf * 8;
        qf[kc].w[p] = *(const u32*)(qrow + kb);
      }
  }

  float mrow[8], lrow[8];
  FragC o[8];
#pragma unroll
  for (int r = 0; r < 8; r++) { mrow[r] = -1e30f; lrow[r] = 0.0f; }
#pragma unroll
  for (int dt = 0; dt < 8; dt++)
#pragma unroll
    for (int j = 0; j < 8; j++) o[dt].f[j] = 0.0f;

  const float scale = rsqrtf((float)dk);
  const int nkt = (q0 + QT) / 32;           // causal: keys < q0+QT

  for (int kt = 0; kt < nkt; kt++) {
    const int kb0 = kt * 32;
    // K tile 32x128 -> Ks[key][d] via async global->LDS (512 16B chunks)
#pragma unroll
    for (int j = 0; j < 2; j++) {
      int ci = tid + j * 256;
      int key = ci >> 4, c = (ci & 15) * 8;
      async_ld_b128(&Ks[key][c], Kh + (size_t)(kb0 + key) * D + c);
    }
    // V tile 32x128 -> Vs[d][key] transposed via regs (256 chunks of 4d x 4key)
    uint2 vreg[4];
    {
      int kq = tid >> 5, nc = (tid & 31) * 4;
#pragma unroll
      for (int r = 0; r < 4; r++)
        vreg[r] = *(const uint2*)(Vh + (size_t)(kb0 + kq * 4 + r) * D + nc);
#pragma unroll
      for (int c2 = 0; c2 < 4; c2++) {
        uint2 st;
        st.x = (u32)((const u16*)&vreg[0])[c2] | ((u32)((const u16*)&vreg[1])[c2] << 16);
        st.y = (u32)((const u16*)&vreg[2])[c2] | ((u32)((const u16*)&vreg[3])[c2] << 16);
        *(uint2*)(&Vs[nc + c2][kq * 4]) = st;
      }
    }
    wait_async();
    __syncthreads();

    // wave-uniform skip: this wave's rows are all below this key tile
    if (kb0 <= qw + 15) {
      // prefetch next K/V tiles
      if (kt + 1 < nkt) {
        size_t nb = (size_t)(kb0 + 32 + lane) * D + (size_t)(wid & 3) * 32;
        __builtin_prefetch(Kh + nb, 0, 0);
        __builtin_prefetch(Vh + nb, 0, 0);
      }

      // scores: preload all 8 B-fragments, then 8 WMMAs
      FragA kf[2][4];
#pragma unroll
      for (int nt = 0; nt < 2; nt++) {
        const int key = nt * 16 + l16;
#pragma unroll
        for (int kc = 0; kc < 4; kc++)
#pragma unroll
          for (int p = 0; p < 8; p++) {
            int db = kc * 32 + (p < 4 ? 2 * p : 8 + 2 * p) + hf * 8;     // B frag: table order over d
            kf[nt][kc].w[p] = *(const u32*)(&Ks[key][db]);
          }
      }
      FragC sc[2];
#pragma unroll
      for (int nt = 0; nt < 2; nt++) {
#pragma unroll
        for (int j = 0; j < 8; j++) sc[nt].f[j] = 0.0f;
#pragma unroll
        for (int kc = 0; kc < 4; kc++)
          sc[nt].m = wmma_bf16(qf[kc].m, kf[nt][kc].m, sc[nt].m);
      }

      // causal mask + online softmax per row; stage P into per-wave LDS (bf16)
#pragma unroll
      for (int r = 0; r < 8; r++) {
        const int qr = qw + r + hf * 8;
        float s0 = sc[0].f[r] * scale; if (kb0 + l16 > qr)      s0 = -1e30f;
        float s1 = sc[1].f[r] * scale; if (kb0 + 16 + l16 > qr) s1 = -1e30f;
        float mt = halfmax16(fmaxf(s0, s1));
        float mn = fmaxf(mrow[r], mt);
        float corr = __expf(mrow[r] - mn);
        float p0 = __expf(s0 - mn), p1 = __expf(s1 - mn);
        float rs = halfsum16(p0 + p1);
        lrow[r] = lrow[r] * corr + rs;
        mrow[r] = mn;
#pragma unroll
        for (int dt = 0; dt < 8; dt++) o[dt].f[r] *= corr;
        Ps[wid][r + hf * 8][l16]      = f2bf(p0);
        Ps[wid][r + hf * 8][16 + l16] = f2bf(p1);
      }

      // P (16x32) as A-fragment; preload 8 V B-fragments; 8 WMMAs
      FragA pa;
#pragma unroll
      for (int p = 0; p < 8; p++) {
        int kb = (p < 4 ? 2 * p : 8 + 2 * p) + hf * 8;
        pa.w[p] = *(const u32*)(&Ps[wid][l16][kb]);
      }
      FragA vb[8];
#pragma unroll
      for (int dt = 0; dt < 8; dt++) {
        const int d = dt * 16 + l16;
#pragma unroll
        for (int p = 0; p < 8; p++) {
          int kk = (p < 4 ? 2 * p : 8 + 2 * p) + hf * 8;               // B frag: table order over keys
          vb[dt].w[p] = *(const u32*)(&Vs[d][kk]);
        }
      }
#pragma unroll
      for (int dt = 0; dt < 8; dt++)
        o[dt].m = wmma_bf16(pa.m, vb[dt].m, o[dt].m);
    }
    __syncthreads();
  }

  // normalize + store bf16 (B,S,D) for final projection GEMM
#pragma unroll
  for (int r = 0; r < 8; r++) {
    float inv = 1.0f / lrow[r];
    size_t row = (size_t)(qw + r + hf * 8) * D;
#pragma unroll
    for (int dt = 0; dt < 8; dt++)
      Oh[row + dt * 16 + l16] = f2bf(o[dt].f[r] * inv);
  }
}

// ---------- launch ----------
extern "C" void kernel_launch(void* const* d_in, const int* in_sizes, int n_in,
                              void* d_out, int out_size, void* d_ws, size_t ws_size,
                              hipStream_t stream) {
  (void)in_sizes; (void)n_in; (void)out_size; (void)ws_size;
  constexpr int Bb = 4, S = 2048, D = 2048, H = 16, DK = 128;
  constexpr int M = Bb * S;                 // 8192
  const float* x  = (const float*)d_in[0];
  const float* Wq = (const float*)d_in[1];
  const float* Wk = (const float*)d_in[2];
  const float* Wv = (const float*)d_in[3];
  const float* Wo = (const float*)d_in[4];

  char* ws = (char*)d_ws;
  const size_t xbBytes = (size_t)M * D * 2;       // 32 MB
  const size_t wBytes  = (size_t)D * D * 2;       // 8 MB
  u16* xb  = (u16*)(ws);                 ws += xbBytes;
  u16* wqb = (u16*)(ws);                 ws += wBytes;
  u16* wkb = (u16*)(ws);                 ws += wBytes;
  u16* wvb = (u16*)(ws);                 ws += wBytes;
  u16* wob = (u16*)(ws);                 ws += wBytes;
  u16* Qb  = (u16*)(ws);                 ws += xbBytes;
  u16* Kb  = (u16*)(ws);                 ws += xbBytes;
  u16* Vb  = (u16*)(ws);                 ws += xbBytes;
  u16* Ab  = (u16*)(ws);                 ws += xbBytes;   // attention output (bf16)

  // 1. fp32 -> bf16
  {
    int n4 = (M * D) / 4;
    cvt_f32_bf16<<<(n4 + 255) / 256, 256, 0, stream>>>((const float4*)x, (uint2*)xb, n4);
    int w4 = (D * D) / 4;
    dim3 g((w4 + 255) / 256);
    cvt_f32_bf16<<<g, 256, 0, stream>>>((const float4*)Wq, (uint2*)wqb, w4);
    cvt_f32_bf16<<<g, 256, 0, stream>>>((const float4*)Wk, (uint2*)wkb, w4);
    cvt_f32_bf16<<<g, 256, 0, stream>>>((const float4*)Wv, (uint2*)wvb, w4);
    cvt_f32_bf16<<<g, 256, 0, stream>>>((const float4*)Wo, (uint2*)wob, w4);
  }

  // 2. QKV projections (bf16 out)
  {
    dim3 grid(D / BN, M / BM);   // (16, 64)
    gemm_bf16_kernel<true><<<grid, 256, 0, stream>>>(xb, wqb, Qb, D, D);
    gemm_bf16_kernel<true><<<grid, 256, 0, stream>>>(xb, wkb, Kb, D, D);
    gemm_bf16_kernel<true><<<grid, 256, 0, stream>>>(xb, wvb, Vb, D, D);
  }

  // 3. RoPE on Q, K
  {
    int total = M * (D / 2);
    rope_kernel<<<(total + 255) / 256, 256, 0, stream>>>(Qb, Kb, M, S, D, DK);
  }

  // 4. attention
  {
    dim3 grid(Bb * H * (S / QT));   // 1024
    attn_kernel<<<grid, 256, 0, stream>>>(Qb, Kb, Vb, Ab, S, H, DK);
  }

  // 5. output projection (fp32 out)
  {
    dim3 grid(D / BN, M / BM);
    gemm_bf16_kernel<false><<<grid, 256, 0, stream>>>(Ab, wob, d_out, D, D);
  }
}
